// E_GraphSAGE_53171695124548
// MI455X (gfx1250) — matrix operand order repaired and checked by reading, not compiled, not run on hardware
//
#include <hip/hip_runtime.h>

typedef __attribute__((ext_vector_type(16))) _Float16 v16h;
typedef __attribute__((ext_vector_type(8)))  float    v8f;

#define HIDDEN 128
#define LPAD   8     // +8 halves (16B) per LDS row: keeps b128 alignment, kills bank conflicts

// ---------------------------------------------------------------------------
// WMMA fragment loaders, contiguous k order (k = 16h + s) for both A and B (wave32).
// A (16x32 f16, MxK): lane L<16 -> M=L, elems 0..15 = K0..15
//                     lane L>=16 -> M=L-16, elems 0..15 = K16..31
// ---------------------------------------------------------------------------
__device__ inline v16h frag_a(const _Float16* p, int lda, int lane) {
    int m  = lane & 15;
    int kb = (lane & 16);                // 0 or 16
    const _Float16* r = p + m * lda + kb;
    v16h a;
#pragma unroll
    for (int i = 0; i < 16; ++i) a[i] = r[i];
    return a;
}

// B (32x16 f16, KxN) from N-major (transposed) weights: row n has K contiguous.
// lanes 0..15 hold K=0..15 (col n=lane), lanes 16..31 hold K=16..31.
__device__ inline v16h frag_b(const _Float16* p /* + ncol0*ldb + kk */, int ldb, int lane) {
    int n  = lane & 15;
    int kb = (lane & 16);                // 0 or 16
    const _Float16* r = p + n * ldb + kb;
    v16h b;
#pragma unroll
    for (int i = 0; i < 16; ++i) b[i] = r[i];
    return b;
}

__device__ inline v8f wmma_f16(v16h a, v16h b, v8f c) {
    return __builtin_amdgcn_wmma_f32_16x16x32_f16(false, a, false, b, (short)0, c,
                                                  false, false);
}

// ---------------------------------------------------------------------------
// Utility kernels
// ---------------------------------------------------------------------------
__global__ void k_zero(float* __restrict__ p, long long n) {
    long long i = (long long)blockIdx.x * blockDim.x + threadIdx.x;
    if (i < n) p[i] = 0.0f;
}

// W [K x N] f32 row-major  ->  Wt [N x K] f16 (K contiguous per output column)
__global__ void k_transpose_w(const float* __restrict__ W, _Float16* __restrict__ Wt,
                              int K, int N) {
    int t = blockIdx.x * blockDim.x + threadIdx.x;
    if (t >= K * N) return;
    int k = t / N, n = t - k * N;
    Wt[(long long)n * K + k] = (_Float16)W[t];
}

// ---------------------------------------------------------------------------
// Edge scatter (mean aggregation numerators + counts) via f32 atomics (L2-res)
// ---------------------------------------------------------------------------
__global__ void k_scatter1(const float* __restrict__ x, const float* __restrict__ ea,
                           const long long* __restrict__ src, const long long* __restrict__ dst,
                           float* __restrict__ agg1, float* __restrict__ cnt,
                           long long total /* E*81 */) {
    long long t = (long long)blockIdx.x * blockDim.x + threadIdx.x;
    if (t >= total) return;
    long long e = t / 81;
    int j = (int)(t - e * 81);
    int s = (int)src[e], d = (int)dst[e];
    if (j < 16)
        atomicAdd(&agg1[(long long)d * 80 + j], x[(long long)s * 16 + j]);
    else if (j < 80)
        atomicAdd(&agg1[(long long)d * 80 + j], ea[e * 64 + (j - 16)]);
    else
        atomicAdd(&cnt[d], 1.0f);
}

__global__ void k_scatter2(const float* __restrict__ h1, const float* __restrict__ ea,
                           const long long* __restrict__ src, const long long* __restrict__ dst,
                           float* __restrict__ agg2, long long total /* E*192 */) {
    long long t = (long long)blockIdx.x * blockDim.x + threadIdx.x;
    if (t >= total) return;
    long long e = t / 192;
    int j = (int)(t - e * 192);
    int s = (int)src[e], d = (int)dst[e];
    float v = (j < 128) ? h1[(long long)s * HIDDEN + j] : ea[e * 64 + (j - 128)];
    atomicAdd(&agg2[(long long)d * 192 + j], v);
}

// ---------------------------------------------------------------------------
// SAGE update: out[node] = ReLU( concat(h_self, agg/cnt) @ W + b ), f16 WMMA.
// One block = 32 nodes; 8 waves, each computes two 16x16 tiles (rows 0-15 and
// 16-31) for one 16-col slice -> B fragments loaded once, used twice.
// ---------------------------------------------------------------------------
__global__ __launch_bounds__(256)
void k_sage_update(const float* __restrict__ hself, int selfdim,
                   const float* __restrict__ agg, int aggdim,
                   const float* __restrict__ cnt,
                   const _Float16* __restrict__ Wt, const float* __restrict__ bias,
                   float* __restrict__ out, int K, int nodes) {
    __shared__ alignas(16) _Float16 A[32 * (320 + LPAD)];
    __shared__ float invc[32];
    const int AS = K + LPAD;
    const int tile = blockIdx.x, tid = threadIdx.x;
    const int base = tile * 32;

    if (tid < 32) {
        int node = base + tid;
        float c = (node < nodes) ? cnt[node] : 1.0f;
        invc[tid] = 1.0f / fmaxf(c, 1.0f);
    }
    __syncthreads();

    for (int idx = tid; idx < 32 * K; idx += 256) {
        int m = idx / K, k = idx - m * K;
        int node = base + m;
        float v = 0.0f;
        if (node < nodes)
            v = (k < selfdim) ? hself[(long long)node * selfdim + k]
                              : agg[(long long)node * aggdim + (k - selfdim)] * invc[m];
        A[m * AS + k] = (_Float16)v;
    }
    __syncthreads();

    const int w = tid >> 5, lane = tid & 31;
    const int ncol0 = w * 16;
    v8f acc0 = {}, acc1 = {};
    for (int kk = 0; kk < K; kk += 32) {
        v16h b  = frag_b(Wt + (long long)ncol0 * K + kk, K, lane);
        v16h a0 = frag_a(&A[kk], AS, lane);
        v16h a1 = frag_a(&A[16 * AS + kk], AS, lane);
        acc0 = wmma_f16(a0, b, acc0);
        acc1 = wmma_f16(a1, b, acc1);
    }
    const int col = ncol0 + (lane & 15);
    const float bv = bias[col];
    const int mo = (lane >> 4) << 3;
#pragma unroll
    for (int r = 0; r < 8; ++r) {
        int n0 = base + mo + r;
        int n1 = n0 + 16;
        if (n0 < nodes) {
            float v = acc0[r] + bv;
            out[(long long)n0 * HIDDEN + col] = v > 0.0f ? v : 0.0f;
        }
        if (n1 < nodes) {
            float v = acc1[r] + bv;
            out[(long long)n1 * HIDDEN + col] = v > 0.0f ? v : 0.0f;
        }
    }
}

// ---------------------------------------------------------------------------
// Classifier: per 32-edge tile, fused 3-layer MLP on WMMA.
//   rep[32x320] @ Wc1 -> ReLU -> @ Wc2[128x64] -> ReLU -> . Wc3[64] -> out
// ---------------------------------------------------------------------------
#define AS1 (320 + LPAD)   // A panel row stride (halves): 164 DW % 64 = 36 -> no conflicts
#define Z1S (128 + LPAD)   // Z1 row stride (halves): 68 DW % 64 = 4 -> no conflicts
#define Z2S 65             // Z2 row stride (f32): coprime with 64 banks

__global__ __launch_bounds__(256)
void k_classifier(const float* __restrict__ h2, const float* __restrict__ ea,
                  const long long* __restrict__ srcI, const long long* __restrict__ dstI,
                  const _Float16* __restrict__ Wc1t, const float* __restrict__ bc1,
                  const _Float16* __restrict__ Wc2t, const float* __restrict__ bc2,
                  const float* __restrict__ Wc3, const float* __restrict__ bc3,
                  float* __restrict__ out, int n_edges) {
    __shared__ alignas(16) _Float16 A[32 * AS1];
    __shared__ alignas(16) _Float16 Z1[32 * Z1S];
    __shared__ float Z2[32 * Z2S];
    __shared__ int ssrc[32], sdst[32];
    const int tile = blockIdx.x, tid = threadIdx.x;
    const int ebase = tile * 32;

    if (tid < 32) {
        int e = ebase + tid;
        int ec = e < n_edges ? e : (n_edges - 1);
        ssrc[tid] = (int)srcI[ec];
        sdst[tid] = (int)dstI[ec];
    }
    __syncthreads();

    for (int idx = tid; idx < 32 * 320; idx += 256) {
        int m = idx / 320, k = idx - m * 320;
        int e = ebase + m;
        float v = 0.0f;
        if (e < n_edges) {
            if (k < 128)       v = h2[(long long)ssrc[m] * HIDDEN + k];
            else if (k < 256)  v = h2[(long long)sdst[m] * HIDDEN + (k - 128)];
            else               v = ea[(long long)e * 64 + (k - 256)];
        }
        A[m * AS1 + k] = (_Float16)v;
    }
    __syncthreads();

    const int w = tid >> 5, lane = tid & 31;
    {   // layer 1: 32x320 @ 320x128; wave w -> cols [w*16, w*16+16), both row halves
        const int ncol0 = w * 16;
        v8f acc0 = {}, acc1 = {};
        for (int kk = 0; kk < 320; kk += 32) {
            v16h b  = frag_b(Wc1t + (long long)ncol0 * 320 + kk, 320, lane);
            v16h a0 = frag_a(&A[kk], AS1, lane);
            v16h a1 = frag_a(&A[16 * AS1 + kk], AS1, lane);
            acc0 = wmma_f16(a0, b, acc0);
            acc1 = wmma_f16(a1, b, acc1);
        }
        const int col = ncol0 + (lane & 15);
        const float bv = bc1[col];
        const int mo = (lane >> 4) << 3;
#pragma unroll
        for (int r = 0; r < 8; ++r) {
            float v0 = acc0[r] + bv;
            float v1 = acc1[r] + bv;
            Z1[(mo + r) * Z1S + col]      = (_Float16)(v0 > 0.0f ? v0 : 0.0f);
            Z1[(16 + mo + r) * Z1S + col] = (_Float16)(v1 > 0.0f ? v1 : 0.0f);
        }
    }
    __syncthreads();

    {   // layer 2: 32x128 @ 128x64 -> 8 tiles of 16x16, exactly one per wave
        const int mh    = (w >> 2) * 16;      // row half: 0 or 16
        const int ncol0 = (w & 3) * 16;       // col tile in [0,64)
        v8f acc = {};
        for (int kk = 0; kk < 128; kk += 32) {
            v16h b = frag_b(Wc2t + (long long)ncol0 * 128 + kk, 128, lane);
            v16h a = frag_a(&Z1[mh * Z1S + kk], Z1S, lane);
            acc = wmma_f16(a, b, acc);
        }
        const int col = ncol0 + (lane & 15);
        const float bv = bc2[col];
        const int mo = (lane >> 4) << 3;
#pragma unroll
        for (int r = 0; r < 8; ++r) {
            float v = acc[r] + bv;
            Z2[(mh + mo + r) * Z2S + col] = v > 0.0f ? v : 0.0f;
        }
    }
    __syncthreads();

    if (tid < 32) {  // layer 3: 32 edges, one full wave; dot with Wc3[64]
        int e = ebase + tid;
        if (e < n_edges) {
            float s = bc3[0];
#pragma unroll 8
            for (int k = 0; k < 64; ++k) s += Z2[tid * Z2S + k] * Wc3[k];
            out[e] = s;
        }
    }
}

// ---------------------------------------------------------------------------
// Launch
// ---------------------------------------------------------------------------
extern "C" void kernel_launch(void* const* d_in, const int* in_sizes, int n_in,
                              void* d_out, int out_size, void* d_ws, size_t ws_size,
                              hipStream_t stream) {
    const float*     x    = (const float*)d_in[0];
    const long long* ei   = (const long long*)d_in[1];
    const float*     ea   = (const float*)d_in[2];
    const float*     W1   = (const float*)d_in[3];
    const float*     b1   = (const float*)d_in[4];
    const float*     W2   = (const float*)d_in[5];
    const float*     b2   = (const float*)d_in[6];
    const float*     Wc1  = (const float*)d_in[7];
    const float*     bc1  = (const float*)d_in[8];
    const float*     Wc2  = (const float*)d_in[9];
    const float*     bc2  = (const float*)d_in[10];
    const float*     Wc3  = (const float*)d_in[11];
    const float*     bc3  = (const float*)d_in[12];
    float* out = (float*)d_out;

    const int n_nodes = in_sizes[0] / 16;   // NODE_DIM=16
    const int n_edges = in_sizes[2] / 64;   // EDGE_DIM=64
    const long long N = n_nodes, E = n_edges;
    const long long* src = ei;
    const long long* dst = ei + E;

    // ---- workspace carve-up (f32 then f16 weight copies) ----
    float* agg1 = (float*)d_ws;            // N*80  (+ cnt contiguous after)
    float* cnt  = agg1 + N * 80;           // N
    float* h1   = cnt + N;                 // N*128
    float* agg2 = h1 + N * 128;            // N*192
    float* h2   = agg2 + N * 192;          // N*128
    _Float16* W1t  = (_Float16*)(h2 + N * 128);  // 128 x 96
    _Float16* W2t  = W1t + 128 * 96;             // 128 x 320
    _Float16* Wc1t = W2t + 128 * 320;            // 128 x 320
    _Float16* Wc2t = Wc1t + 128 * 320;           // 64 x 128

    const int B = 256;

    // zero aggregation buffers (+counts)
    {
        long long z1 = N * 81;   // agg1 + cnt contiguous
        k_zero<<<(int)((z1 + B - 1) / B), B, 0, stream>>>(agg1, z1);
        long long z2 = N * 192;
        k_zero<<<(int)((z2 + B - 1) / B), B, 0, stream>>>(agg2, z2);
    }

    // weight conversion/transposition to f16 N-major
    k_transpose_w<<<(96 * 128 + B - 1) / B, B, 0, stream>>>(W1, W1t, 96, 128);
    k_transpose_w<<<(320 * 128 + B - 1) / B, B, 0, stream>>>(W2, W2t, 320, 128);
    k_transpose_w<<<(320 * 128 + B - 1) / B, B, 0, stream>>>(Wc1, Wc1t, 320, 128);
    k_transpose_w<<<(128 * 64 + B - 1) / B, B, 0, stream>>>(Wc2, Wc2t, 128, 64);

    // layer 1 aggregate + update
    {
        long long total = E * 81;
        k_scatter1<<<(int)((total + B - 1) / B), B, 0, stream>>>(x, ea, src, dst,
                                                                 agg1, cnt, total);
    }
    {
        int tiles = (n_nodes + 31) / 32;
        k_sage_update<<<tiles, B, 0, stream>>>(x, 16, agg1, 80, cnt, W1t, b1,
                                               h1, 96, n_nodes);
    }

    // layer 2 aggregate + update
    {
        long long total = E * 192;
        k_scatter2<<<(int)((total + B - 1) / B), B, 0, stream>>>(h1, ea, src, dst,
                                                                 agg2, total);
    }
    {
        int tiles = (n_nodes + 31) / 32;
        k_sage_update<<<tiles, B, 0, stream>>>(h1, 128, agg2, 192, cnt, W2t, b2,
                                               h2, 320, n_nodes);
    }

    // fused per-edge classifier MLP
    {
        int tiles = (n_edges + 31) / 32;
        k_classifier<<<tiles, B, 0, stream>>>(h2, ea, src, dst,
                                              Wc1t, bc1, Wc2t, bc2, Wc3, bc3,
                                              out, n_edges);
    }
}
